// CausalSelfAttention_18124761989585
// MI455X (gfx1250) — matrix operand order repaired and checked by reading, not compiled, not run on hardware
//
#include <hip/hip_runtime.h>

typedef __attribute__((ext_vector_type(16))) __bf16 v16bf;
typedef __attribute__((ext_vector_type(8)))  float  v8f;
typedef unsigned short u16;
typedef unsigned int   u32;

namespace {
constexpr int Bb = 4;          // batch
constexpr int T  = 2048;       // seq len
constexpr int C  = 768;        // embed
constexpr int H  = 12;         // heads
constexpr int Dh = 64;         // head dim
constexpr int M1 = Bb * T;     // 8192
constexpr int N1 = 3 * C;      // 2304
constexpr int BM = 64, BN = 128, BK = 32;
constexpr int SA = BK + 8;     // LDS row stride (ushorts): 40 -> 80B rows, 16B aligned
}

__device__ __forceinline__ u16 f32_bf16(float f) {
  u32 u = __builtin_bit_cast(u32, f);
  u = (u + 0x7FFFu + ((u >> 16) & 1u)) >> 16;   // round-to-nearest-even
  return (u16)u;
}

struct Frag {
  union { v16bf v; u32 d[8]; };
};

__device__ __forceinline__ v8f wmma_bf16(const Frag& a, const Frag& b, v8f c) {
  // D = A(16x32 bf16) * B(32x16 bf16) + C(16x16 f32)
  return __builtin_amdgcn_wmma_f32_16x16x32_bf16(
      false, a.v, false, b.v, (short)0, c, false, false);
}

// CDNA5 async global->LDS copy (ASYNCcnt-tracked, no VGPR staging).
// VDST = LDS byte address (low 32 bits of flat shared pointer == LDS offset
// per the ISA aperture mapping), VADDR = 64-bit global address, GV mode.
__device__ __forceinline__ void async_copy_b128(const u16* gsrc, u32 lds_off) {
  asm volatile("global_load_async_to_lds_b128 %0, %1, off"
               :: "v"(lds_off), "v"(gsrc) : "memory");
}
template <int N>
__device__ __forceinline__ void wait_async() {
  asm volatile("s_wait_asynccnt %0" :: "i"(N) : "memory");
}

// ---------------------------------------------------------------- converts
__global__ void k_convert(const float* __restrict__ src, u16* __restrict__ dst, int n) {
  int i = blockIdx.x * blockDim.x + threadIdx.x;
  int stride = gridDim.x * blockDim.x;
  for (; i < n; i += stride) dst[i] = f32_bf16(src[i]);
}

// f32 [rows][cols] -> bf16 transposed [cols][rows], LDS-tiled 32x32
__global__ void __launch_bounds__(1024)
k_convT(const float* __restrict__ src, u16* __restrict__ dst, int rows, int cols) {
  __shared__ float tile[32][33];
  const int r0 = blockIdx.x * 32, c0 = blockIdx.y * 32;
  const int tx = threadIdx.x, ty = threadIdx.y;
  tile[ty][tx] = src[(size_t)(r0 + ty) * cols + c0 + tx];
  __syncthreads();
  dst[(size_t)(c0 + ty) * rows + r0 + tx] = f32_bf16(tile[tx][ty]);
}

// ---------------------------------------------------------------- GEMM core
// 64x128 block tile, 128 threads (4 waves), each wave 32x64 (2x4 WMMA tiles).
// Both tiles staged with async global->LDS copies into double-buffered LDS;
// next tile's DMA is issued before waiting so it overlaps the WMMAs.
// Bt is the pre-transposed weight matrix [N][K] so all copies are contiguous.
__device__ __forceinline__ void issue_tile(const u16* __restrict__ A,
                                           const u16* __restrict__ Bt,
                                           int lda, int ldbt,
                                           int row0, int col0, int k0,
                                           u16* Asub, u16* BsubT, int tid) {
  // A tile 64x32 bf16: 256 x 16B chunks, 2 per thread
#pragma unroll
  for (int i = 0; i < 2; ++i) {
    int u = tid + i * 128;
    int row = u >> 2;                // 4 chunks per 64B row
    int cq  = (u & 3) << 3;          // chunk start (ushorts)
    async_copy_b128(&A[(size_t)(row0 + row) * lda + k0 + cq],
                    (u32)(size_t)&Asub[row * SA + cq]);
  }
  // B tile (transposed source) 128x32 bf16: 512 x 16B chunks, 4 per thread
#pragma unroll
  for (int i = 0; i < 4; ++i) {
    int u = tid + i * 128;
    int n  = u >> 2;
    int cq = (u & 3) << 3;
    async_copy_b128(&Bt[(size_t)(col0 + n) * ldbt + k0 + cq],
                    (u32)(size_t)&BsubT[n * SA + cq]);
  }
}

__device__ __forceinline__ void gemm_tile(const u16* __restrict__ A,
                                          const u16* __restrict__ Bt,
                                          int lda, int ldbt,
                                          int row0, int col0, int Ksz,
                                          v8f acc[2][4],
                                          u16 (*Asub)[BM * SA],
                                          u16 (*BsubT)[BN * SA]) {
  const int tid  = threadIdx.x;
  const int lane = tid & 31;
  const int wave = tid >> 5;
  const int wm = wave >> 1, wn = wave & 1;
  const int hlf = lane >> 4;       // lane half (ISA 16-bit operand layouts)
  const int ml  = lane & 15;       // A row within tile
  const int nl  = lane & 15;       // B col within tile

  for (int tm = 0; tm < 2; ++tm)
    for (int tn = 0; tn < 4; ++tn)
      for (int r = 0; r < 8; ++r) acc[tm][tn][r] = 0.0f;

  const int nk = Ksz / BK;
  issue_tile(A, Bt, lda, ldbt, row0, col0, 0, Asub[0], BsubT[0], tid);

#pragma unroll 1
  for (int it = 0; it < nk; ++it) {
    const int cur = it & 1;
    if (it + 1 < nk) {   // issue-ahead: overlap next tile's DMA with compute
      issue_tile(A, Bt, lda, ldbt, row0, col0, (it + 1) * BK,
                 Asub[cur ^ 1], BsubT[cur ^ 1], tid);
      wait_async<6>();   // in-order: <=6 outstanding => current tile resident
    } else {
      wait_async<0>();
    }
    __syncthreads();

    Frag af[2], bfr[4];
#pragma unroll
    for (int tm = 0; tm < 2; ++tm) {
      const u16* base = &Asub[cur][(wm * 32 + tm * 16 + ml) * SA];
#pragma unroll
      for (int r = 0; r < 8; ++r) {
        int k = ((r >> 2) << 4) + (hlf << 3) + ((r & 3) << 1); // ISA 16-bit A 16x32
        af[tm].d[r] = *(const u32*)&base[k];
      }
    }
#pragma unroll
    for (int tn = 0; tn < 4; ++tn) {
      const u16* base = &BsubT[cur][(wn * 64 + tn * 16 + nl) * SA];
#pragma unroll
      for (int r = 0; r < 8; ++r) {
        int k = ((r >> 2) << 4) + (hlf << 3) + ((r & 3) << 1); // ISA 16-bit B 32x16
        bfr[tn].d[r] = *(const u32*)&base[k];
      }
    }
#pragma unroll
    for (int tm = 0; tm < 2; ++tm)
#pragma unroll
      for (int tn = 0; tn < 4; ++tn)
        acc[tm][tn] = wmma_bf16(af[tm], bfr[tn], acc[tm][tn]);
    __syncthreads();   // all waves done reading before buffer is re-filled
  }
}

// ------------------------------------------------- GEMM1: QKV proj + split
// qkv = x @ W_attn + b ; epilogue scatters into Q[bh][T][D], K[bh][T][D],
// Vt[bh][D][T] (V pre-transposed for the PV matmul's B-fragment loads).
__global__ void __launch_bounds__(128)
k_qkv(const u16* __restrict__ xb, const u16* __restrict__ wt,
      const float* __restrict__ bias,
      u16* __restrict__ Qb, u16* __restrict__ Kb, u16* __restrict__ Vt) {
  __shared__ __align__(16) u16 Asub[2][BM * SA];
  __shared__ __align__(16) u16 BsubT[2][BN * SA];
  const int row0 = blockIdx.x * BM, col0 = blockIdx.y * BN;
  v8f acc[2][4];
  gemm_tile(xb, wt, C, C, row0, col0, C, acc, Asub, BsubT);

  const int lane = threadIdx.x & 31, wave = threadIdx.x >> 5;
  const int wm = wave >> 1, wn = wave & 1;
  const int hlf = lane >> 4, nl = lane & 15;
#pragma unroll
  for (int tm = 0; tm < 2; ++tm)
#pragma unroll
    for (int tn = 0; tn < 4; ++tn) {
      int gc = col0 + wn * 64 + tn * 16 + nl;
      int part = gc / C, cc = gc % C;
      int hh = cc >> 6, d = cc & 63;
      float bv = bias[gc];
#pragma unroll
      for (int r = 0; r < 8; ++r) {
        int gr = row0 + wm * 32 + tm * 16 + r + (hlf << 3); // C layout: m=r+8*half
        int bidx = gr >> 11, t = gr & 2047;
        u16 bits = f32_bf16(acc[tm][tn][r] + bv);
        size_t bh = (size_t)(bidx * H + hh);
        if (part == 0)      Qb[(bh * T + t) * Dh + d] = bits;
        else if (part == 1) Kb[(bh * T + t) * Dh + d] = bits;
        else                Vt[(bh * Dh + d) * T + t] = bits;
      }
    }
}

// ------------------------------------------------- flash attention
// One wave per 16-row query tile. Q fragments resident; online softmax;
// P staged through LDS to convert C-layout -> A-layout for the PV WMMA.
__global__ void __launch_bounds__(128)
k_attn(const u16* __restrict__ Qb, const u16* __restrict__ Kb,
       const u16* __restrict__ Vt, u16* __restrict__ yb) {
  __shared__ u16 Pl[4][16 * 36];   // per-wave P staging, stride 36
  const int bh   = blockIdx.x;     // 0..47
  const int wave = threadIdx.x >> 5;
  const int lane = threadIdx.x & 31;
  const int qt   = blockIdx.y * 4 + wave;
  const int q0   = qt * 16;
  const int hlf  = lane >> 4;
  const int ml   = lane & 15;
  const int nl   = lane & 15;

  // Q fragments: A layout 16x32, two fragments cover D=64
  Frag qf[2];
  {
    const u16* qbase = Qb + ((size_t)bh * T + q0 + ml) * Dh;
#pragma unroll
    for (int f = 0; f < 2; ++f)
#pragma unroll
      for (int r = 0; r < 8; ++r) {
        int k = f * 32 + ((r >> 2) << 4) + (hlf << 3) + ((r & 3) << 1);
        qf[f].d[r] = *(const u32*)&qbase[k];
      }
  }

  float mrun[8], lrun[8];
  v8f acc[4];
#pragma unroll
  for (int r = 0; r < 8; ++r) { mrun[r] = -3.0e38f; lrun[r] = 0.0f; }
#pragma unroll
  for (int dt = 0; dt < 4; ++dt)
#pragma unroll
    for (int r = 0; r < 8; ++r) acc[dt][r] = 0.0f;

  const int kend = q0 + 16;                     // causal bound (exclusive)
#pragma unroll 1
  for (int k0 = 0; k0 < kend; k0 += 32) {
    // ---- scores = Q K^T for 16x32 tile (two 16-wide n-subtiles)
    v8f s[2];
#pragma unroll
    for (int j = 0; j < 2; ++j) {
#pragma unroll
      for (int r = 0; r < 8; ++r) s[j][r] = 0.0f;
      Frag kf0, kf1;
      const u16* kb = Kb + ((size_t)bh * T + k0 + j * 16 + nl) * Dh;
#pragma unroll
      for (int r = 0; r < 8; ++r) {
        int d0 = ((r >> 2) << 4) + (hlf << 3) + ((r & 3) << 1);
        kf0.d[r] = *(const u32*)&kb[d0];        // d = 0..31
        kf1.d[r] = *(const u32*)&kb[32 + d0];   // d = 32..63
      }
      s[j] = wmma_bf16(qf[0], kf0, s[j]);
      s[j] = wmma_bf16(qf[1], kf1, s[j]);
    }
    // ---- scale + causal mask
#pragma unroll
    for (int j = 0; j < 2; ++j)
#pragma unroll
      for (int r = 0; r < 8; ++r) {
        int q = q0 + r + (hlf << 3);
        int kk = k0 + j * 16 + nl;
        float v = s[j][r] * 0.125f;             // 1/sqrt(64)
        s[j][r] = (kk <= q) ? v : -3.0e38f;
      }
    // ---- online softmax (rows live within a 16-lane half)
    float mnew[8];
#pragma unroll
    for (int r = 0; r < 8; ++r) {
      float mx = fmaxf(s[0][r], s[1][r]);
      mx = fmaxf(mx, __shfl_xor(mx, 1, 32));
      mx = fmaxf(mx, __shfl_xor(mx, 2, 32));
      mx = fmaxf(mx, __shfl_xor(mx, 4, 32));
      mx = fmaxf(mx, __shfl_xor(mx, 8, 32));
      mnew[r] = fmaxf(mrun[r], mx);
    }
#pragma unroll
    for (int r = 0; r < 8; ++r) {
      float sc = __expf(mrun[r] - mnew[r]);
      float p0 = __expf(s[0][r] - mnew[r]);
      float p1 = __expf(s[1][r] - mnew[r]);
      float rs = p0 + p1;
      rs += __shfl_xor(rs, 1, 32);
      rs += __shfl_xor(rs, 2, 32);
      rs += __shfl_xor(rs, 4, 32);
      rs += __shfl_xor(rs, 8, 32);
      lrun[r] = lrun[r] * sc + rs;
      mrun[r] = mnew[r];
#pragma unroll
      for (int dt = 0; dt < 4; ++dt) acc[dt][r] *= sc;
      int m = r + (hlf << 3);
      Pl[wave][m * 36 + nl]      = f32_bf16(p0);
      Pl[wave][m * 36 + 16 + nl] = f32_bf16(p1);
    }
    asm volatile("s_wait_dscnt 0" ::: "memory");  // wave-local LDS WAR->RAW
    // ---- P as A-fragment (16x32)
    Frag pf;
    {
      const u16* pbase = &Pl[wave][ml * 36];
#pragma unroll
      for (int r = 0; r < 8; ++r) {
        int k = ((r >> 2) << 4) + (hlf << 3) + ((r & 3) << 1);
        pf.d[r] = *(const u32*)&pbase[k];
      }
    }
    // ---- acc += P V  (V pre-transposed: rows are d, contiguous along T)
#pragma unroll
    for (int dt = 0; dt < 4; ++dt) {
      Frag vf;
      const u16* vb = Vt + ((size_t)bh * Dh + dt * 16 + nl) * T + k0;
#pragma unroll
      for (int r = 0; r < 8; ++r) {
        int kk = ((r >> 2) << 4) + (hlf << 3) + ((r & 3) << 1);
        vf.d[r] = *(const u32*)&vb[kk];
      }
      acc[dt] = wmma_bf16(pf, vf, acc[dt]);
    }
  }

  // ---- normalize, write y back to [B*T][C] bf16 for the output GEMM
  const int b = bh / H, hh = bh % H;
#pragma unroll
  for (int dt = 0; dt < 4; ++dt)
#pragma unroll
    for (int r = 0; r < 8; ++r) {
      int t = q0 + r + (hlf << 3);
      int col = hh * 64 + dt * 16 + nl;
      float v = acc[dt][r] / lrun[r];
      yb[((size_t)(b * T + t)) * C + col] = f32_bf16(v);
    }
}

// ------------------------------------------------- GEMM2: output projection
__global__ void __launch_bounds__(128)
k_proj(const u16* __restrict__ yb, const u16* __restrict__ wt,
       const float* __restrict__ bias, float* __restrict__ out) {
  __shared__ __align__(16) u16 Asub[2][BM * SA];
  __shared__ __align__(16) u16 BsubT[2][BN * SA];
  const int row0 = blockIdx.x * BM, col0 = blockIdx.y * BN;
  v8f acc[2][4];
  gemm_tile(yb, wt, C, C, row0, col0, C, acc, Asub, BsubT);

  const int lane = threadIdx.x & 31, wave = threadIdx.x >> 5;
  const int wm = wave >> 1, wn = wave & 1;
  const int hlf = lane >> 4, nl = lane & 15;
#pragma unroll
  for (int tm = 0; tm < 2; ++tm)
#pragma unroll
    for (int tn = 0; tn < 4; ++tn) {
      int gc = col0 + wn * 64 + tn * 16 + nl;
      float bv = bias[gc];
#pragma unroll
      for (int r = 0; r < 8; ++r) {
        int gr = row0 + wm * 32 + tm * 16 + r + (hlf << 3);
        out[(size_t)gr * C + gc] = acc[tm][tn][r] + bv;
      }
    }
}

// ----------------------------------------------------------------- launch
extern "C" void kernel_launch(void* const* d_in, const int* in_sizes, int n_in,
                              void* d_out, int out_size, void* d_ws, size_t ws_size,
                              hipStream_t stream) {
  const float* x      = (const float*)d_in[0];
  const float* W_attn = (const float*)d_in[1];
  const float* b_attn = (const float*)d_in[2];
  const float* W_o    = (const float*)d_in[3];
  const float* b_o    = (const float*)d_in[4];
  float* out = (float*)d_out;

  size_t off = 0;
  char* base = (char*)d_ws;
  auto carve = [&](size_t bytes) -> void* {
    void* p = base + off;
    off += (bytes + 255) & ~(size_t)255;
    return p;
  };
  u16* xb    = (u16*)carve((size_t)M1 * C  * 2);
  u16* wqkvT = (u16*)carve((size_t)C  * N1 * 2);  // [N1][C] bf16
  u16* woT   = (u16*)carve((size_t)C  * C  * 2);  // [C][C] bf16 (transposed)
  u16* Qb    = (u16*)carve((size_t)Bb * H * T * Dh * 2);
  u16* Kb    = (u16*)carve((size_t)Bb * H * T * Dh * 2);
  u16* Vt    = (u16*)carve((size_t)Bb * H * T * Dh * 2);
  u16* yb    = (u16*)carve((size_t)M1 * C  * 2);
  (void)ws_size; (void)in_sizes; (void)n_in; (void)out_size;

  k_convert<<<512, 256, 0, stream>>>(x, xb, M1 * C);
  k_convT<<<dim3(C / 32, N1 / 32), dim3(32, 32), 0, stream>>>(W_attn, wqkvT, C, N1);
  k_convT<<<dim3(C / 32, C / 32), dim3(32, 32), 0, stream>>>(W_o, woT, C, C);

  k_qkv<<<dim3(M1 / BM, N1 / BN), 128, 0, stream>>>(xb, wqkvT, b_attn, Qb, Kb, Vt);
  k_attn<<<dim3(Bb * H, (T / 16) / 4), 128, 0, stream>>>(Qb, Kb, Vt, yb);
  k_proj<<<dim3(M1 / BM, C / BN), 128, 0, stream>>>(yb, woT, b_o, out);
}
